// ModuleCorrelation_41970420416706
// MI455X (gfx1250) — matrix operand order repaired and checked by reading, not compiled, not run on hardware
//
#include <hip/hip_runtime.h>

// PWC-Net correlation for MI455X (gfx1250): per-(b,y) workgroup, WMMA f16->f32
// cross-correlation GEMM, diagonal extraction of the 9x9 displacement window.
// v2: pipelined B-fragment DS loads, branchless diagonal scatter (junk row),
//     store/stage phase fusion (2 barriers per dy).

typedef _Float16 v8h  __attribute__((ext_vector_type(8)));
typedef _Float16 v16h __attribute__((ext_vector_type(16)));
typedef _Float16 h4   __attribute__((ext_vector_type(4)));
typedef float    v8f  __attribute__((ext_vector_type(8)));
typedef float    f4   __attribute__((ext_vector_type(4)));

namespace {
constexpr int kB = 8;
constexpr int kC = 128;
constexpr int kH = 128;
constexpr int kW = 128;
constexpr int SSTR = 136;   // LDS row stride in halves ([x][c] image, c contiguous)
constexpr int OSTR = 132;   // LDS out-tile row stride in floats
}

union V16 { v16h v; v8h h[2]; };

__global__ __launch_bounds__(256) void corr81_wmma_kernel(
    const float* __restrict__ first,
    const float* __restrict__ second,
    float* __restrict__ out) {
  __shared__ _Float16 ldsA[128 * SSTR];   // 34.0 KB  first row, [x][c] f16
  __shared__ _Float16 ldsB[144 * SSTR];   // 38.3 KB  second row, +/-8 x zero borders
  __shared__ float    ldsO[10 * OSTR];    //  5.2 KB  rows 0..8 = dx, row 9 = junk sink

  const int t    = threadIdx.x;
  const int lane = t & 31;
  const int wave = t >> 5;           // 8 waves (wave32)
  const int y    = blockIdx.x;
  const int b    = blockIdx.y;
  const int x0   = wave * 16;        // this wave's M-tile of x positions

  const size_t plane = (size_t)kH * kW;
  const float* fbase = first  + ((size_t)b * kC) * plane + (size_t)y * kW;
  const float* sbase = second + ((size_t)b * kC) * plane;

  // Staging map: thread handles c-rows [c0, c0+4) and x columns 4*(t&7)+32*i.
  const int c0 = (t >> 3) * 4;
  const int xb = (t & 7) * 4;

  // stage `second` row sy (or zeros) into ldsB rows [8,136)
  auto stageB = [&](int sy) {
    if ((unsigned)sy < (unsigned)kH) {
      const float* srow = sbase + (size_t)sy * kW;
#pragma unroll
      for (int i = 0; i < 4; ++i) {
        const int x = xb + 32 * i;
        f4 r[4];
#pragma unroll
        for (int cc = 0; cc < 4; ++cc)
          r[cc] = *(const f4*)(srow + (size_t)(c0 + cc) * plane + x);
#pragma unroll
        for (int xx = 0; xx < 4; ++xx) {
          h4 v;
          v.x = (_Float16)r[0][xx];
          v.y = (_Float16)r[1][xx];
          v.z = (_Float16)r[2][xx];
          v.w = (_Float16)r[3][xx];
          *(h4*)&ldsB[(size_t)(x + xx + 8) * SSTR + c0] = v;
        }
      }
      if ((unsigned)(sy + 1) < (unsigned)kH)   // warm next dy row in L2
        __builtin_prefetch(sbase + (size_t)c0 * plane + (size_t)(sy + 1) * kW + xb, 0, 1);
    } else {
      const h4 z = {};
#pragma unroll
      for (int i = 0; i < 4; ++i) {
        const int x = xb + 32 * i;
#pragma unroll
        for (int xx = 0; xx < 4; ++xx)
          *(h4*)&ldsB[(size_t)(x + xx + 8) * SSTR + c0] = z;
      }
    }
  };

  // ---- stage `first` row y into ldsA (transpose to [x][c], f32->f16) ----
#pragma unroll
  for (int i = 0; i < 4; ++i) {
    const int x = xb + 32 * i;
    f4 r[4];
#pragma unroll
    for (int cc = 0; cc < 4; ++cc)
      r[cc] = *(const f4*)(fbase + (size_t)(c0 + cc) * plane + x);
#pragma unroll
    for (int xx = 0; xx < 4; ++xx) {
      h4 v;
      v.x = (_Float16)r[0][xx];
      v.y = (_Float16)r[1][xx];
      v.z = (_Float16)r[2][xx];
      v.w = (_Float16)r[3][xx];
      *(h4*)&ldsA[(size_t)(x + xx) * SSTR + c0] = v;   // ds_store_b64
    }
  }
  // zero the x-borders of ldsB once (rows 0..7 and 136..143)
  {
    int* bz0 = (int*)&ldsB[0];
    int* bz1 = (int*)&ldsB[136 * SSTR];
    for (int idx = t; idx < (8 * SSTR) / 2; idx += 256) { bz0[idx] = 0; bz1[idx] = 0; }
  }
  // stage first `second` row (dy = 0 -> sy = y - 4)
  stageB(y - 4);
  __syncthreads();

  // ---- A fragments (16x32 f16, ISA 7.12.2 lane layout), resident in VGPRs ----
  // lane<16: m=lane, K = k0+0..7 | k0+16..23 ; lane>=16: m=lane-16, K = k0+8..15 | k0+24..31
  v16h afrag[4];
  {
    const int m    = lane & 15;
    const int asel = (lane >> 4) * 8;
#pragma unroll
    for (int kc = 0; kc < 4; ++kc) {
      const _Float16* p = &ldsA[(size_t)(x0 + m) * SSTR + kc * 32 + asel];
      V16 u;
      u.h[0] = *(const v8h*)(p);
      u.h[1] = *(const v8h*)(p + 16);
      afrag[kc] = u.v;
    }
  }

  const int   n     = lane & 15;        // B-fragment column / D-fragment N
  const int   hi    = lane >> 4;
  const int   bksel = hi * 8;           // lane>=16 holds K=k0+8..15 | k0+24..31
  const float invC  = 1.0f / (float)kC;

  const _Float16* bbase0 = &ldsB[(size_t)(x0 + n) * SSTR + bksel];        // tile0 rows x0+n
  const _Float16* bbase1 = bbase0 + (size_t)16 * SSTR;                    // tile1 rows x0+16+n

  for (int dy = 0; dy < 9; ++dy) {
    // ---- two 16x16x128 GEMM tiles, depth-2 pipelined B loads ----
    v8f acc0 = {0.f, 0.f, 0.f, 0.f, 0.f, 0.f, 0.f, 0.f};
    v8f acc1 = {0.f, 0.f, 0.f, 0.f, 0.f, 0.f, 0.f, 0.f};
    V16 bf0[2], bf1[2];
    bf0[0].h[0] = *(const v8h*)(bbase0);
    bf0[0].h[1] = *(const v8h*)(bbase0 + 16);
    bf1[0].h[0] = *(const v8h*)(bbase1);
    bf1[0].h[1] = *(const v8h*)(bbase1 + 16);
#pragma unroll
    for (int kc = 0; kc < 4; ++kc) {
      const int cur = kc & 1;
      const int nxt = cur ^ 1;
      if (kc < 3) {   // prefetch next K-chunk fragments while current WMMAs run
        const _Float16* p0 = bbase0 + (kc + 1) * 32;
        const _Float16* p1 = bbase1 + (kc + 1) * 32;
        bf0[nxt].h[0] = *(const v8h*)(p0);
        bf0[nxt].h[1] = *(const v8h*)(p0 + 16);
        bf1[nxt].h[0] = *(const v8h*)(p1);
        bf1[nxt].h[1] = *(const v8h*)(p1 + 16);
      }
      acc0 = __builtin_amdgcn_wmma_f32_16x16x32_f16(false, afrag[kc], false, bf0[cur].v,
                                                    (short)0, acc0, false, false);
      acc1 = __builtin_amdgcn_wmma_f32_16x16x32_f16(false, afrag[kc], false, bf1[cur].v,
                                                    (short)0, acc1, false, false);
    }

    // ---- branchless diagonal extraction into ldsO (row 9 = junk sink) ----
    // D layout: VGPR r, lane l -> (m = r + 8*(l>=16), n = l%16)
    // tile0: sx = x0-8+n -> dx = n-m-4 ; tile1: sx = x0+8+n -> dx = n-m+12
#pragma unroll
    for (int r = 0; r < 8; ++r) {
      const int m    = r + 8 * hi;
      const int d0   = n - m - 4;
      const int d1   = n - m + 12;
      const int row0 = ((unsigned)d0 <= 8u) ? d0 : 9;   // v_cndmask, no exec branch
      const int row1 = ((unsigned)d1 <= 8u) ? d1 : 9;
      ldsO[row0 * OSTR + x0 + m] = acc0[r] * invC;
      ldsO[row1 * OSTR + x0 + m] = acc1[r] * invC;
    }
    __syncthreads();

    // ---- fused phase: coalesced output store + stage next second-row ----
    for (int j = t; j < 288; j += 256) {
      const int dx = j >> 5;
      const int xq = (j & 31) * 4;
      f4 v = *(const f4*)&ldsO[dx * OSTR + xq];
      const size_t o = (((size_t)b * 81 + (size_t)(dy * 9 + dx)) * kH + y) * kW + xq;
      *(f4*)&out[o] = v;
    }
    if (dy < 8) stageB(y + dy - 3);    // sy for dy+1
    __syncthreads();   // ldsB ready for next compute; ldsO free for rewrite
  }
}

extern "C" void kernel_launch(void* const* d_in, const int* in_sizes, int n_in,
                              void* d_out, int out_size, void* d_ws, size_t ws_size,
                              hipStream_t stream) {
  (void)in_sizes; (void)n_in; (void)out_size; (void)d_ws; (void)ws_size;
  const float* first  = (const float*)d_in[0];
  const float* second = (const float*)d_in[1];
  float*       out    = (float*)d_out;
  dim3 grid(kH, kB);   // one workgroup per (y, b)
  corr81_wmma_kernel<<<grid, 256, 0, stream>>>(first, second, out);
}
